// NER_CRF_Model_40175124087526
// MI455X (gfx1250) — matrix operand order repaired and checked by reading, not compiled, not run on hardware
//
#include <hip/hip_runtime.h>
#include <hip/hip_bf16.h>

typedef __attribute__((ext_vector_type(16))) __bf16 v16bf;
typedef __attribute__((ext_vector_type(8)))  float  v8f;
typedef __attribute__((ext_vector_type(4)))  int    v4i;

union Frag16 {
  v16bf bf;
  v4i   q[2];
};

__device__ __forceinline__ unsigned short f2bf(float f) {
  unsigned int u = __float_as_uint(f);
  u += 0x7fffu + ((u >> 16) & 1u);           // round-to-nearest-even
  return (unsigned short)(u >> 16);
}

__device__ __forceinline__ v8f wmma_bf16(const Frag16& a, const Frag16& b, v8f c) {
  return __builtin_amdgcn_wmma_f32_16x16x32_bf16(false, a.bf, false, b.bf,
                                                 (short)0, c, false, false);
}

// ---- gfx1250 async global->LDS copy path (fallback: sync copy) -------------
#ifdef __has_builtin
#if __has_builtin(__builtin_amdgcn_global_load_async_to_lds_b128) && \
    __has_builtin(__builtin_amdgcn_s_wait_asynccnt)
#define USE_ASYNC_LDS 1
#endif
#endif

typedef __attribute__((address_space(1))) v4i GV4;   // global 16B vector
typedef __attribute__((address_space(3))) v4i LV4;   // LDS 16B vector

__device__ __forceinline__ void stage16(const unsigned short* g, unsigned short* l) {
#ifdef USE_ASYNC_LDS
  __builtin_amdgcn_global_load_async_to_lds_b128((GV4*)g, (LV4*)l, 0, 0);
#else
  *reinterpret_cast<uint4*>(l) = *reinterpret_cast<const uint4*>(g);
#endif
}
__device__ __forceinline__ void stage_wait() {
#ifdef USE_ASYNC_LDS
  __builtin_amdgcn_s_wait_asynccnt(0);
#endif
}

// ---------------------------------------------------------------------------
// One-time weight convert + transpose: W[K][N] fp32 -> Wt[N][K] bf16
// ---------------------------------------------------------------------------
__global__ __launch_bounds__(256) void wcvt_kernel(
    const float* __restrict__ W, unsigned short* __restrict__ Wt, int K, int N)
{
  __shared__ unsigned short tile[64 * 65];
  const int n0 = blockIdx.x * 64;
  const int k0 = blockIdx.y * 64;
  const int tid = threadIdx.x;
#pragma unroll
  for (int i = 0; i < 4; ++i) {
    int lin = tid + i * 256;
    int kk = lin >> 4;
    int nn = (lin & 15) << 2;
    float4 v = *reinterpret_cast<const float4*>(W + (size_t)(k0 + kk) * N + n0 + nn);
    unsigned short* p = &tile[kk * 65 + nn];
    p[0] = f2bf(v.x); p[1] = f2bf(v.y); p[2] = f2bf(v.z); p[3] = f2bf(v.w);
  }
  __syncthreads();
#pragma unroll
  for (int i = 0; i < 4; ++i) {
    int lin = tid + i * 256;
    int nn = lin >> 4;
    int kk = (lin & 15) << 2;
    unsigned long long pk =
        (unsigned long long)tile[(kk + 0) * 65 + nn] |
        ((unsigned long long)tile[(kk + 1) * 65 + nn] << 16) |
        ((unsigned long long)tile[(kk + 2) * 65 + nn] << 32) |
        ((unsigned long long)tile[(kk + 3) * 65 + nn] << 48);
    *reinterpret_cast<unsigned long long*>(Wt + (size_t)(n0 + nn) * K + k0 + kk) = pk;
  }
}

// ---------------------------------------------------------------------------
// WMMA GEMM:  C[M,N] = act(A[M,K] @ Wt[N,K]^T + bias)
// bf16 A and pre-transposed bf16 weights; staging is pure b128 copies
// (async when available) with double-buffered LDS tiles.
// ---------------------------------------------------------------------------
template<int FUSE_GELU, int OUT_BF16>
__global__ __launch_bounds__(256) void gemm_wmma_kernel(
    const unsigned short* __restrict__ A,   // [M][K] bf16
    const unsigned short* __restrict__ Wt,  // [N][K] bf16
    const float* __restrict__ bias,
    void* __restrict__ Cout, int K, int N)
{
  __shared__ unsigned short As[2][128 * 40];
  __shared__ unsigned short Bs[2][128 * 40];
  const int tid  = threadIdx.x;
  const int lane = tid & 31;
  const int wave = tid >> 5;
  const int n0 = blockIdx.x * 128;
  const int m0 = blockIdx.y * 128;
  const int wm = wave & 3;   // 4 row groups of 32
  const int wn = wave >> 2;  // 2 col groups of 64

  v8f acc[2][4];
#pragma unroll
  for (int i = 0; i < 2; ++i)
#pragma unroll
    for (int j = 0; j < 4; ++j) acc[i][j] = {};

  const int akb = (lane >> 4) << 3;   // ISA 16-bit A-layout k-base
  const int bkb = (lane >> 4) << 3;   // B k-base, same order as A
  const int srow = tid >> 2;          // staging: rows srow, srow+64
  const int sc   = (tid & 3) << 3;    // 8-element (16B) chunks

  auto stageAB = [&](int k0, int buf) {
    const unsigned short* ag = A + (size_t)(m0 + srow) * K + k0 + sc;
    stage16(ag,                  &As[buf][srow * 40 + sc]);
    stage16(ag + (size_t)64 * K, &As[buf][(srow + 64) * 40 + sc]);
    const unsigned short* bg = Wt + (size_t)(n0 + srow) * K + k0 + sc;
    stage16(bg,                  &Bs[buf][srow * 40 + sc]);
    stage16(bg + (size_t)64 * K, &Bs[buf][(srow + 64) * 40 + sc]);
  };

  const int nIter = K >> 5;
  stageAB(0, 0);
  stage_wait();
  __syncthreads();

  for (int it = 0; it < nIter; ++it) {
    const int cur = it & 1;
    if (it + 1 < nIter) stageAB((it + 1) << 5, cur ^ 1);   // prefetch next tile

    const int arow = wm * 32 + (lane & 15);
    Frag16 a[2];
#pragma unroll
    for (int mi = 0; mi < 2; ++mi) {
      const unsigned short* p = &As[cur][(arow + mi * 16) * 40 + akb];
      a[mi].q[0] = *reinterpret_cast<const v4i*>(p);        // K = akb..akb+7
      a[mi].q[1] = *reinterpret_cast<const v4i*>(p + 16);   // K = akb+16..+23
    }
    const int bcol = wn * 64 + (lane & 15);
#pragma unroll
    for (int ni = 0; ni < 4; ++ni) {
      Frag16 b;
      const unsigned short* p = &Bs[cur][(bcol + ni * 16) * 40 + bkb];
      b.q[0] = *reinterpret_cast<const v4i*>(p);
      b.q[1] = *reinterpret_cast<const v4i*>(p + 16);
#pragma unroll
      for (int mi = 0; mi < 2; ++mi)
        acc[mi][ni] = wmma_bf16(a[mi], b, acc[mi][ni]);
    }
    stage_wait();
    __syncthreads();
  }

  const int cn   = lane & 15;
  const int rsel = (lane >> 4) << 3;
#pragma unroll
  for (int ni = 0; ni < 4; ++ni) {
    const int col = n0 + wn * 64 + ni * 16 + cn;
    const float bv = bias[col];
#pragma unroll
    for (int mi = 0; mi < 2; ++mi) {
#pragma unroll
      for (int v = 0; v < 8; ++v) {
        const int row = m0 + wm * 32 + mi * 16 + rsel + v;
        float x = acc[mi][ni][v] + bv;
        if (FUSE_GELU) {
          const float c0 = 0.7978845608028654f;
          float t = tanhf(c0 * (x + 0.044715f * x * x * x));
          x = 0.5f * x * (1.0f + t);
        }
        if (OUT_BF16)
          ((unsigned short*)Cout)[(size_t)row * N + col] = f2bf(x);
        else
          ((float*)Cout)[(size_t)row * N + col] = x;
      }
    }
  }
}

// ---------------------------------------------------------------------------
// Embedding: out[token] = word_emb[id] + pos_emb[s]   (fp32)
// ---------------------------------------------------------------------------
__global__ __launch_bounds__(256) void embed_kernel(
    const int* __restrict__ ids, const float* __restrict__ we,
    const float* __restrict__ pe, float* __restrict__ out)
{
  const int row = blockIdx.x;
  const int s = row & 255;
  const int id = ids[row];
  const size_t base = (size_t)row * 768;
#pragma unroll
  for (int i = 0; i < 3; ++i) {
    int c = threadIdx.x + i * 256;
    out[base + c] = we[(size_t)id * 768 + c] + pe[(size_t)s * 768 + c];
  }
}

// ---------------------------------------------------------------------------
// LayerNorm (+ optional residual); writes fp32 master AND bf16 GEMM copy
// ---------------------------------------------------------------------------
__global__ __launch_bounds__(256) void ln_kernel(
    const float* __restrict__ inp, const float* __restrict__ resid,
    const float* __restrict__ s, const float* __restrict__ b,
    float* __restrict__ out, unsigned short* __restrict__ out_bf)
{
  __shared__ float red[8];
  const int tid = threadIdx.x;
  const int lane = tid & 31, wave = tid >> 5;
  const size_t base = (size_t)blockIdx.x * 768;
  float vals[3];
  float sum = 0.f;
#pragma unroll
  for (int i = 0; i < 3; ++i) {
    int c = tid + i * 256;
    float x = inp[base + c];
    if (resid) x += resid[base + c];
    vals[i] = x; sum += x;
  }
#pragma unroll
  for (int o = 16; o > 0; o >>= 1) sum += __shfl_xor(sum, o, 32);
  if (lane == 0) red[wave] = sum;
  __syncthreads();
  float mean = 0.f;
#pragma unroll
  for (int i = 0; i < 8; ++i) mean += red[i];
  mean *= (1.0f / 768.0f);
  __syncthreads();
  float vs = 0.f;
#pragma unroll
  for (int i = 0; i < 3; ++i) { float d = vals[i] - mean; vs += d * d; }
#pragma unroll
  for (int o = 16; o > 0; o >>= 1) vs += __shfl_xor(vs, o, 32);
  if (lane == 0) red[wave] = vs;
  __syncthreads();
  float var = 0.f;
#pragma unroll
  for (int i = 0; i < 8; ++i) var += red[i];
  var *= (1.0f / 768.0f);
  const float rstd = rsqrtf(var + 1e-12f);
#pragma unroll
  for (int i = 0; i < 3; ++i) {
    int c = tid + i * 256;
    float y = (vals[i] - mean) * rstd * s[c] + b[c];
    out[base + c] = y;
    out_bf[base + c] = f2bf(y);
  }
}

// ---------------------------------------------------------------------------
// Attention: one block per (batch, head, 32-query tile). bf16 QKV input.
// Scores and P@V via WMMA; Q/K staged with async copies; softmax in-wave.
// ---------------------------------------------------------------------------
__global__ __launch_bounds__(256) void attn_kernel(
    const unsigned short* __restrict__ qkv, const int* __restrict__ amask,
    unsigned short* __restrict__ attn)
{
  __shared__ unsigned short KV[64 * 72];    // Kt[key][d] pass1 / VtT[d][key] pass2
  __shared__ unsigned short Qt[32 * 72];
  __shared__ unsigned short Pm[32 * 264];
  __shared__ float          Sc[32 * 260];
  __shared__ float          biasL[256];

  const int blk = blockIdx.x;               // bb*96 + h*8 + qt
  const int qt = blk & 7;
  const int h  = (blk >> 3) % 12;
  const int bb = blk / 96;
  const int q0 = qt * 32;
  const int tid = threadIdx.x, lane = tid & 31, wave = tid >> 5;
  const size_t base = (size_t)bb * 256 * 2304;

  biasL[tid] = (1.0f - (float)amask[bb * 256 + tid]) * -1e9f;

  // stage Q tile 32x64 (1 x b128 per thread)
  {
    int q = tid >> 3;
    int c = (tid & 7) << 3;
    stage16(qkv + base + (size_t)(q0 + q) * 2304 + h * 64 + c, &Qt[q * 72 + c]);
  }

  const int wm   = wave & 1;
  const int wn   = wave >> 1;
  const int cn   = lane & 15;
  const int rsel = (lane >> 4) << 3;
  const int akb  = (lane >> 4) << 3;
  const int bkb  = (lane >> 4) << 3;

  // ---- pass 1: scores = Q @ K^T * 1/8 + bias ----
  for (int kc = 0; kc < 4; ++kc) {
    if (kc) __syncthreads();
#pragma unroll
    for (int i = 0; i < 2; ++i) {           // stage 64 keys of K
      int lin = tid + i * 256;
      int key = lin >> 3;
      int c = (lin & 7) << 3;
      stage16(qkv + base + (size_t)(kc * 64 + key) * 2304 + 768 + h * 64 + c,
              &KV[key * 72 + c]);
    }
    stage_wait();
    __syncthreads();

    v8f s8 = {};
    const int arow = wm * 16 + (lane & 15);
    const int bcol = wn * 16 + (lane & 15);
#pragma unroll
    for (int kk = 0; kk < 64; kk += 32) {
      Frag16 a, b;
      const unsigned short* pa = &Qt[arow * 72 + kk + akb];
      a.q[0] = *reinterpret_cast<const v4i*>(pa);
      a.q[1] = *reinterpret_cast<const v4i*>(pa + 16);
      const unsigned short* pb = &KV[bcol * 72 + kk + bkb];
      b.q[0] = *reinterpret_cast<const v4i*>(pb);
      b.q[1] = *reinterpret_cast<const v4i*>(pb + 16);
      s8 = wmma_bf16(a, b, s8);
    }
#pragma unroll
    for (int v = 0; v < 8; ++v) {
      int row = wm * 16 + rsel + v;
      int col = kc * 64 + wn * 16 + cn;
      Sc[row * 260 + col] = s8[v] * 0.125f + biasL[col];
    }
  }
  __syncthreads();

  // ---- softmax, 4 rows per wave, 8 cols per lane ----
#pragma unroll
  for (int r = 0; r < 4; ++r) {
    int row = wave * 4 + r;
    float x[8];
    float mx = -3.0e38f;
#pragma unroll
    for (int i = 0; i < 8; ++i) { x[i] = Sc[row * 260 + lane + i * 32]; mx = fmaxf(mx, x[i]); }
#pragma unroll
    for (int o = 16; o > 0; o >>= 1) mx = fmaxf(mx, __shfl_xor(mx, o, 32));
    float sm = 0.f;
#pragma unroll
    for (int i = 0; i < 8; ++i) { x[i] = expf(x[i] - mx); sm += x[i]; }
#pragma unroll
    for (int o = 16; o > 0; o >>= 1) sm += __shfl_xor(sm, o, 32);
    float inv = 1.0f / sm;
#pragma unroll
    for (int i = 0; i < 8; ++i) Pm[row * 264 + lane + i * 32] = f2bf(x[i] * inv);
  }

  // ---- pass 2: O = P @ V over key chunks ----
  v8f o8 = {};
  const int arow2 = wm * 16 + (lane & 15);
  const int dcol  = wn * 16 + (lane & 15);
  for (int kc = 0; kc < 4; ++kc) {
    __syncthreads();
#pragma unroll
    for (int i = 0; i < 2; ++i) {           // stage V chunk transposed VtT[d][key]
      int lin = tid + i * 256;
      int key = lin >> 3;
      int c = (lin & 7) << 3;
      uint4 v = *reinterpret_cast<const uint4*>(
          qkv + base + (size_t)(kc * 64 + key) * 2304 + 1536 + h * 64 + c);
      const unsigned short* pv = reinterpret_cast<const unsigned short*>(&v);
#pragma unroll
      for (int j = 0; j < 8; ++j) KV[(c + j) * 72 + key] = pv[j];
    }
    __syncthreads();
#pragma unroll
    for (int kk = 0; kk < 64; kk += 32) {
      Frag16 a, b;
      const unsigned short* pa = &Pm[arow2 * 264 + kc * 64 + kk + akb];
      a.q[0] = *reinterpret_cast<const v4i*>(pa);
      a.q[1] = *reinterpret_cast<const v4i*>(pa + 16);
      const unsigned short* pb = &KV[dcol * 72 + kk + bkb];
      b.q[0] = *reinterpret_cast<const v4i*>(pb);
      b.q[1] = *reinterpret_cast<const v4i*>(pb + 16);
      o8 = wmma_bf16(a, b, o8);
    }
  }
#pragma unroll
  for (int v = 0; v < 8; ++v) {
    int q = q0 + wm * 16 + rsel + v;
    int d = wn * 16 + cn;
    attn[((size_t)bb * 256 + q) * 768 + h * 64 + d] = f2bf(o8[v]);
  }
}

// ---------------------------------------------------------------------------
// Classifier: logits[token,9] = x @ Wcls + bcls  (one wave per token)
// ---------------------------------------------------------------------------
__global__ __launch_bounds__(32) void cls_kernel(
    const float* __restrict__ x, const float* __restrict__ Wc,
    const float* __restrict__ bc, float* __restrict__ logits)
{
  const int row = blockIdx.x;
  const int lane = threadIdx.x;
  const float* xr = x + (size_t)row * 768;
  for (int t = 0; t < 9; ++t) {
    float s = 0.f;
    for (int i = lane; i < 768; i += 32) s += xr[i] * Wc[i * 9 + t];
#pragma unroll
    for (int o = 16; o > 0; o >>= 1) s += __shfl_xor(s, o, 32);
    if (lane == 0) logits[(size_t)row * 9 + t] = s + bc[t];
  }
}

// ---------------------------------------------------------------------------
// CRF NLL: one wave, lanes 0..15 handle one batch each
// ---------------------------------------------------------------------------
__global__ __launch_bounds__(32) void crf_kernel(
    const float* __restrict__ logits, const int* __restrict__ labels,
    const float* __restrict__ cs, const float* __restrict__ ce,
    const float* __restrict__ ct, float* __restrict__ loss)
{
  __shared__ float trans[81];
  __shared__ float lred[16];
  const int lane = threadIdx.x;
  for (int i = lane; i < 81; i += 32) trans[i] = ct[i];
  __syncthreads();
  if (lane < 16) {
    const int b = lane;
    const float* lg = logits + (size_t)b * 256 * 9;
    const int* lb = labels + (size_t)b * 256;
    int tag_prev = (lb[0] == -100) ? 0 : lb[0];   // mask[:,0] forced true
    float num = cs[tag_prev] + lg[tag_prev];
    int cnt = 1;
    for (int s = 1; s < 256; ++s) {
      int l = lb[s];
      bool mk = (l != -100);
      int tg = mk ? l : 0;
      if (mk) { num += trans[tag_prev * 9 + tg] + lg[s * 9 + tg]; cnt += 1; }
      tag_prev = tg;
    }
    int lt_idx = cnt - 1;                         // seq_end = sum(mask)-1
    int lraw = lb[lt_idx];
    int last_tag = (lt_idx == 0) ? ((lb[0] == -100) ? 0 : lb[0])
                                 : ((lraw == -100) ? 0 : lraw);
    num += ce[last_tag];

    float sc[9];
#pragma unroll
    for (int j = 0; j < 9; ++j) sc[j] = cs[j] + lg[j];
    for (int s = 1; s < 256; ++s) {
      bool mk = (lb[s] != -100);
      float nxt[9];
#pragma unroll
      for (int j = 0; j < 9; ++j) {
        float mx = -3.0e38f;
#pragma unroll
        for (int i = 0; i < 9; ++i) mx = fmaxf(mx, sc[i] + trans[i * 9 + j]);
        float sm = 0.f;
#pragma unroll
        for (int i = 0; i < 9; ++i) sm += expf(sc[i] + trans[i * 9 + j] - mx);
        nxt[j] = mx + logf(sm) + lg[s * 9 + j];
      }
#pragma unroll
      for (int j = 0; j < 9; ++j) sc[j] = mk ? nxt[j] : sc[j];
    }
    float mx = -3.0e38f;
#pragma unroll
    for (int j = 0; j < 9; ++j) mx = fmaxf(mx, sc[j] + ce[j]);
    float sm = 0.f;
#pragma unroll
    for (int j = 0; j < 9; ++j) sm += expf(sc[j] + ce[j] - mx);
    lred[b] = -(num - (mx + logf(sm)));
  }
  __syncthreads();
  if (lane == 0) {
    float s = 0.f;
#pragma unroll
    for (int i = 0; i < 16; ++i) s += lred[i];
    loss[0] = s * (1.0f / 16.0f);
  }
}

// ---------------------------------------------------------------------------
extern "C" void kernel_launch(void* const* d_in, const int* in_sizes, int n_in,
                              void* d_out, int out_size, void* d_ws, size_t ws_size,
                              hipStream_t stream) {
  (void)in_sizes; (void)n_in; (void)out_size; (void)ws_size;
  const int*   ids      = (const int*)d_in[0];
  const int*   amask    = (const int*)d_in[1];
  const int*   labels   = (const int*)d_in[2];
  const float* word_emb = (const float*)d_in[3];
  const float* pos_emb  = (const float*)d_in[4];
  const float* eln_s    = (const float*)d_in[5];
  const float* eln_b    = (const float*)d_in[6];
  const float* Wqkv     = (const float*)d_in[7];
  const float* bqkv     = (const float*)d_in[8];
  const float* Wo       = (const float*)d_in[9];
  const float* bo       = (const float*)d_in[10];
  const float* ln1s     = (const float*)d_in[11];
  const float* ln1b     = (const float*)d_in[12];
  const float* Wff1     = (const float*)d_in[13];
  const float* bff1     = (const float*)d_in[14];
  const float* Wff2     = (const float*)d_in[15];
  const float* bff2     = (const float*)d_in[16];
  const float* ln2s     = (const float*)d_in[17];
  const float* ln2b     = (const float*)d_in[18];
  const float* Wcls     = (const float*)d_in[19];
  const float* bcls     = (const float*)d_in[20];
  const float* crf_s    = (const float*)d_in[21];
  const float* crf_e    = (const float*)d_in[22];
  const float* crf_t    = (const float*)d_in[23];

  const int M = 16 * 256;   // 4096 tokens
  char* p = (char*)d_ws;
  auto alloc = [&](size_t bytes) { void* r = p; p += (bytes + 255) & ~(size_t)255; return r; };
  float*          x       = (float*)alloc((size_t)M * 768 * 4);
  float*          t2      = (float*)alloc((size_t)M * 768 * 4);
  unsigned short* x_bf    = (unsigned short*)alloc((size_t)M * 768 * 2);
  unsigned short* qkv_bf  = (unsigned short*)alloc((size_t)M * 2304 * 2);
  unsigned short* attn_bf = (unsigned short*)alloc((size_t)M * 768 * 2);
  unsigned short* t1_bf   = (unsigned short*)alloc((size_t)M * 3072 * 2);
  unsigned short* WqkvT   = (unsigned short*)alloc((size_t)6 * 2304 * 768 * 2);
  unsigned short* WoT     = (unsigned short*)alloc((size_t)6 * 768 * 768 * 2);
  unsigned short* Wff1T   = (unsigned short*)alloc((size_t)6 * 3072 * 768 * 2);
  unsigned short* Wff2T   = (unsigned short*)alloc((size_t)6 * 768 * 3072 * 2);
  float* outf   = (float*)d_out;
  float* logits = outf + 1;

  // one-time (per launch) weight convert+transpose to bf16 Wt[N][K]
  for (int l = 0; l < 6; ++l) {
    wcvt_kernel<<<dim3(2304 / 64, 768 / 64), 256, 0, stream>>>(
        Wqkv + (size_t)l * 768 * 2304, WqkvT + (size_t)l * 2304 * 768, 768, 2304);
    wcvt_kernel<<<dim3(768 / 64, 768 / 64), 256, 0, stream>>>(
        Wo + (size_t)l * 768 * 768, WoT + (size_t)l * 768 * 768, 768, 768);
    wcvt_kernel<<<dim3(3072 / 64, 768 / 64), 256, 0, stream>>>(
        Wff1 + (size_t)l * 768 * 3072, Wff1T + (size_t)l * 3072 * 768, 768, 3072);
    wcvt_kernel<<<dim3(768 / 64, 3072 / 64), 256, 0, stream>>>(
        Wff2 + (size_t)l * 3072 * 768, Wff2T + (size_t)l * 768 * 3072, 3072, 768);
  }

  embed_kernel<<<M, 256, 0, stream>>>(ids, word_emb, pos_emb, t2);
  ln_kernel<<<M, 256, 0, stream>>>(t2, nullptr, eln_s, eln_b, x, x_bf);

  for (int l = 0; l < 6; ++l) {
    gemm_wmma_kernel<0, 1><<<dim3(2304 / 128, M / 128), 256, 0, stream>>>(
        x_bf, WqkvT + (size_t)l * 2304 * 768, bqkv + (size_t)l * 2304, qkv_bf, 768, 2304);
    attn_kernel<<<16 * 12 * 8, 256, 0, stream>>>(qkv_bf, amask, attn_bf);
    gemm_wmma_kernel<0, 0><<<dim3(768 / 128, M / 128), 256, 0, stream>>>(
        attn_bf, WoT + (size_t)l * 768 * 768, bo + (size_t)l * 768, t2, 768, 768);
    ln_kernel<<<M, 256, 0, stream>>>(t2, x, ln1s + (size_t)l * 768, ln1b + (size_t)l * 768, x, x_bf);
    gemm_wmma_kernel<1, 1><<<dim3(3072 / 128, M / 128), 256, 0, stream>>>(
        x_bf, Wff1T + (size_t)l * 3072 * 768, bff1 + (size_t)l * 3072, t1_bf, 768, 3072);
    gemm_wmma_kernel<0, 0><<<dim3(768 / 128, M / 128), 256, 0, stream>>>(
        t1_bf, Wff2T + (size_t)l * 768 * 3072, bff2 + (size_t)l * 768, t2, 3072, 768);
    ln_kernel<<<M, 256, 0, stream>>>(t2, x, ln2s + (size_t)l * 768, ln2b + (size_t)l * 768, x, x_bf);
  }

  cls_kernel<<<M, 32, 0, stream>>>(x, Wcls, bcls, logits);
  crf_kernel<<<1, 32, 0, stream>>>(logits, labels, crf_s, crf_e, crf_t, outf);
}
